// AttentionBlock_73641509257386
// MI455X (gfx1250) — matrix operand order repaired and checked by reading, not compiled, not run on hardware
//
#include <hip/hip_runtime.h>

typedef __attribute__((ext_vector_type(16))) _Float16 v16h;
typedef __attribute__((ext_vector_type(8)))  _Float16 v8h;
typedef __attribute__((ext_vector_type(8)))  float    v8f;

#define BATCH     16
#define DIM       512
#define HEADS     8
#define DIM_HEAD  64
#define SEQ       1024                 // 32*32
#define INNER     (HEADS * DIM_HEAD)   // 512
#define N_QKV     (INNER * 3)          // 1536
#define ATT_SCALE 0.125f               // 64^-0.5

// ---------------------------------------------------------------------------
// WMMA helper (gfx1250: V_WMMA_F32_16X16X32_F16, wave32)
// ---------------------------------------------------------------------------
__device__ __forceinline__ v8f wmma16(v16h a, v16h b, v8f c) {
  return __builtin_amdgcn_wmma_f32_16x16x32_f16(
      false, a, false, b, (short)0, c, false, false);
}

union V16U { v16h v; v8h h[2]; };

// A-matrix 16x32 fragment from row-major [m][k] source, row stride `rs` halves
// (multiple of 8 -> 16B-aligned lane pointers; two b128 loads per lane).
__device__ __forceinline__ v16h frag_a(const _Float16* base, int rs, int lane) {
  const _Float16* p = base + (lane & 15) * rs + ((lane >> 4) << 3);
  V16U u;
  u.h[0] = *(const v8h*)p;
  u.h[1] = *(const v8h*)(p + 16);
  return u.v;
}

// B-matrix 32x16 fragment where memory holds B transposed: M[n][k] row-major.
__device__ __forceinline__ v16h frag_bT(const _Float16* base, int rs, int lane) {
  const _Float16* p = base + (lane & 15) * rs + ((lane >> 4) << 3);
  V16U u;
  u.h[0] = *(const v8h*)p;
  u.h[1] = *(const v8h*)(p + 16);
  return u.v;
}

// ---------------------------------------------------------------------------
// CDNA5 async global->LDS copies (GLOBAL_LOAD_ASYNC_TO_LDS_B128, ASYNCcnt).
// INST_OFFSET is added to BOTH the global and LDS addresses (ISA 15.18.3),
// so contiguous multi-chunk copies reuse one address pair with offset:N.
// LDS operand = low 32 bits of the generic pointer.
// ---------------------------------------------------------------------------
__device__ __forceinline__ void async_cp16(const void* g, void* l) {
  unsigned lds = (unsigned)(uintptr_t)l;
  unsigned long long ga = (unsigned long long)(uintptr_t)g;
  asm volatile("global_load_async_to_lds_b128 %0, %1, off"
               :: "v"(lds), "v"(ga) : "memory");
}
__device__ __forceinline__ void async_cp32(const void* g, void* l) {
  unsigned lds = (unsigned)(uintptr_t)l;
  unsigned long long ga = (unsigned long long)(uintptr_t)g;
  asm volatile("global_load_async_to_lds_b128 %0, %1, off\n\t"
               "global_load_async_to_lds_b128 %0, %1, off offset:16"
               :: "v"(lds), "v"(ga) : "memory");
}
__device__ __forceinline__ void async_cp64(const void* g, void* l) {
  unsigned lds = (unsigned)(uintptr_t)l;
  unsigned long long ga = (unsigned long long)(uintptr_t)g;
  asm volatile("global_load_async_to_lds_b128 %0, %1, off\n\t"
               "global_load_async_to_lds_b128 %0, %1, off offset:16\n\t"
               "global_load_async_to_lds_b128 %0, %1, off offset:32\n\t"
               "global_load_async_to_lds_b128 %0, %1, off offset:48"
               :: "v"(lds), "v"(ga) : "memory");
}
__device__ __forceinline__ void wait_async0() {
  asm volatile("s_wait_asynccnt 0x0" ::: "memory");
}

// ---------------------------------------------------------------------------
// Kernel 0: one-time weight transpose+convert: dst f16 [n][k] <- src f32 [k][n].
// Makes every GEMM B-tile staging a pure contiguous f16 copy (async-to-LDS).
// ---------------------------------------------------------------------------
__global__ __launch_bounds__(256) void wtrans_kernel(
    const float* __restrict__ src, _Float16* __restrict__ dst, int K, int N) {
  int idx = blockIdx.x * 256 + threadIdx.x;
  if (idx < K * N) {
    int k = idx / N, n = idx - k * N;
    dst[(size_t)n * K + k] = (_Float16)src[idx];
  }
}

// ---------------------------------------------------------------------------
// Kernel 1: QKV projection.  64(M) x 128(N) tile, K-step 32, 8 waves x 4 accs.
// A staged from x (f32, strided -> cvt); B staged via async from WpT f16.
// Output scattered to q/k/v f16 [b][h][s][d]; q pre-scaled by ATT_SCALE.
// ---------------------------------------------------------------------------
__global__ __launch_bounds__(256) void qkv_proj_kernel(
    const float* __restrict__ x, const _Float16* __restrict__ WpT,
    const float* __restrict__ bp,
    _Float16* __restrict__ qbuf, _Float16* __restrict__ kbuf,
    _Float16* __restrict__ vbuf) {
  __shared__ __align__(16) _Float16 Al[64][40];    // [s][k]
  __shared__ __align__(16) _Float16 BlT[128][40];  // [n][k]

  const int m0 = blockIdx.x * 64;
  const int n0 = blockIdx.y * 128;
  const int wave = threadIdx.x >> 5, lane = threadIdx.x & 31;
  const int b = m0 >> 10, s_base = m0 & 1023;
  const int tm = wave & 3;
  const int tnb = (wave >> 2) * 4;

  v8f acc[4] = {{}, {}, {}, {}};

  for (int k0 = 0; k0 < DIM; k0 += 32) {
    __syncthreads();
    {   // A tile: x[(b*512+k)*1024+s] (strided f32 -> f16, one b128 LDS store)
      int sI = threadIdx.x & 63;
      int kI = (threadIdx.x >> 6) * 8;
      const float* gx = x + ((size_t)(b * DIM + k0 + kI)) * SEQ + s_base + sI;
      v8h a8;
#pragma unroll
      for (int i = 0; i < 8; ++i) a8[i] = (_Float16)gx[(size_t)i * SEQ];
      *(v8h*)&Al[sI][kI] = a8;
    }
    {   // B tile: pure f16 copy from WpT[n][k] -> async to LDS (2 ops/thread)
      int nI = threadIdx.x & 127;
      int kr = (threadIdx.x >> 7) * 16;
      async_cp32(WpT + (size_t)(n0 + nI) * DIM + k0 + kr, &BlT[nI][kr]);
    }
    wait_async0();
    __syncthreads();

    v16h a = frag_a(&Al[tm * 16][0], 40, lane);
    v16h bf[4];
#pragma unroll
    for (int j = 0; j < 4; ++j)
      bf[j] = frag_bT(&BlT[(tnb + j) * 16][0], 40, lane);
#pragma unroll
    for (int j = 0; j < 4; ++j) acc[j] = wmma16(a, bf[j], acc[j]);
  }

  // Epilogue: tile-uniform head/part/dbase, strided f16 stores.
  const int nlane = lane & 15, hi = lane >> 4;
#pragma unroll
  for (int j = 0; j < 4; ++j) {
    int ncb = n0 + (tnb + j) * 16;
    int h = ncb / 192;
    int part = (ncb - h * 192) >> 6;        // 0=q 1=k 2=v
    int dbase = ncb & 63;
    float bias = bp[ncb + nlane];
    _Float16* dst = (part == 0) ? qbuf : (part == 1) ? kbuf : vbuf;
    float scl = (part == 0) ? ATT_SCALE : 1.0f;
    _Float16* dp = dst +
        (((size_t)(b * HEADS + h)) * SEQ + s_base + tm * 16 + 8 * hi) * DIM_HEAD +
        dbase + nlane;
#pragma unroll
    for (int r = 0; r < 8; ++r)
      dp[(size_t)r * DIM_HEAD] = (_Float16)((acc[j][r] + bias) * scl);
  }
}

// ---------------------------------------------------------------------------
// Kernel 2: flash attention per (b,h).  Grid (128,16), 128 threads (4 waves);
// wave owns 16 query rows; online softmax over 64-key chunks.
// K chunk: pure f16 copy -> async-to-LDS.  V chunk: transposed (VGPR scatter).
// ---------------------------------------------------------------------------
__global__ __launch_bounds__(128) void attn_kernel(
    const _Float16* __restrict__ qbuf, const _Float16* __restrict__ kbuf,
    const _Float16* __restrict__ vbuf, _Float16* __restrict__ resbuf) {
  __shared__ __align__(16) _Float16 Klds[64][64];      // [key][d]
  __shared__ __align__(16) _Float16 VldsT[64][80];     // [d][key]
  __shared__ __align__(16) _Float16 Plds[4][16][64];   // per-wave P

  const int bh = blockIdx.x;                 // b*8 + h
  const int b = bh >> 3, h = bh & 7;
  const int wave = threadIdx.x >> 5, lane = threadIdx.x & 31;
  const int q0 = blockIdx.y * 64 + wave * 16;
  const int nlane = lane & 15, hi = lane >> 4;

  const _Float16* Qb = qbuf + ((size_t)bh * SEQ + q0) * DIM_HEAD;
  const _Float16* Kb = kbuf + (size_t)bh * SEQ * DIM_HEAD;
  const _Float16* Vb = vbuf + (size_t)bh * SEQ * DIM_HEAD;

  v16h qa0 = frag_a(Qb, DIM_HEAD, lane);        // d 0..31
  v16h qa1 = frag_a(Qb + 32, DIM_HEAD, lane);   // d 32..63

  v8f acc[4] = {{}, {}, {}, {}};
  float rmax[8], rsum[8];
#pragma unroll
  for (int r = 0; r < 8; ++r) { rmax[r] = -1e30f; rsum[r] = 0.0f; }

  for (int c0 = 0; c0 < SEQ; c0 += 64) {
    __syncthreads();
    {
      int key = threadIdx.x >> 1, dseg = (threadIdx.x & 1) * 32;
      // K: 64B contiguous copy per thread -> 4 async b128 ops.
      async_cp64(Kb + (size_t)(c0 + key) * DIM_HEAD + dseg, &Klds[key][dseg]);
      // V: transpose into [d][key] (contiguous global reads, u16 LDS scatter).
      const _Float16* gv = Vb + (size_t)(c0 + key) * DIM_HEAD + dseg;
      v8h vv[4];
#pragma unroll
      for (int i = 0; i < 4; ++i) vv[i] = *(const v8h*)(gv + i * 8);
#pragma unroll
      for (int i = 0; i < 32; ++i) VldsT[dseg + i][key] = vv[i >> 3][i & 7];
    }
    wait_async0();
    __syncthreads();

    // Scores: 4 key-tiles x 2 d-halves -> 8 WMMA.
    v8f sc[4] = {{}, {}, {}, {}};
#pragma unroll
    for (int j = 0; j < 4; ++j) {
      v16h kf0 = frag_bT(&Klds[j * 16][0], 64, lane);
      v16h kf1 = frag_bT(&Klds[j * 16][32], 64, lane);
      sc[j] = wmma16(qa0, kf0, sc[j]);
      sc[j] = wmma16(qa1, kf1, sc[j]);
    }

    // Online softmax: row m = r+8*hi lives in one 16-lane group at VGPR r.
    float alpha[8];
#pragma unroll
    for (int r = 0; r < 8; ++r) {
      float v = fmaxf(fmaxf(sc[0][r], sc[1][r]), fmaxf(sc[2][r], sc[3][r]));
#pragma unroll
      for (int m = 8; m >= 1; m >>= 1) v = fmaxf(v, __shfl_xor(v, m, 16));
      float mnew = fmaxf(rmax[r], v);
      alpha[r] = __expf(rmax[r] - mnew);
      rmax[r] = mnew;
      float ps = 0.0f;
#pragma unroll
      for (int j = 0; j < 4; ++j) {
        float p = __expf(sc[j][r] - mnew);
        sc[j][r] = p;
        ps += p;
      }
#pragma unroll
      for (int m = 8; m >= 1; m >>= 1) ps += __shfl_xor(ps, m, 16);
      rsum[r] = rsum[r] * alpha[r] + ps;
    }
#pragma unroll
    for (int t = 0; t < 4; ++t)
#pragma unroll
      for (int r = 0; r < 8; ++r) acc[t][r] *= alpha[r];

    // P (D-layout) -> per-wave LDS -> A-layout fragments (in-wave DS order).
#pragma unroll
    for (int j = 0; j < 4; ++j)
#pragma unroll
      for (int r = 0; r < 8; ++r)
        Plds[wave][r + 8 * hi][j * 16 + nlane] = (_Float16)sc[j][r];
    asm volatile("s_wait_dscnt 0x0" ::: "memory");
    v16h pa0 = frag_a(&Plds[wave][0][0], 64, lane);    // keys 0..31
    v16h pa1 = frag_a(&Plds[wave][0][32], 64, lane);   // keys 32..63

    // O += P(16x64) . V(64x64): 4 d-tiles x 2 k-steps -> 8 WMMA.
#pragma unroll
    for (int t = 0; t < 4; ++t) {
      v16h vf0 = frag_bT(&VldsT[t * 16][0], 80, lane);
      v16h vf1 = frag_bT(&VldsT[t * 16][32], 80, lane);
      acc[t] = wmma16(pa0, vf0, acc[t]);
      acc[t] = wmma16(pa1, vf1, acc[t]);
    }
  }

  // Normalize and store: res[b][s][h*64+d] f16.
  float inv[8];
#pragma unroll
  for (int r = 0; r < 8; ++r) inv[r] = 1.0f / rsum[r];
#pragma unroll
  for (int t = 0; t < 4; ++t) {
    _Float16* rp = resbuf + ((size_t)(b * SEQ + q0 + 8 * hi)) * INNER +
                   h * DIM_HEAD + t * 16 + nlane;
#pragma unroll
    for (int r = 0; r < 8; ++r)
      rp[(size_t)r * INNER] = (_Float16)(acc[t][r] * inv[r]);
  }
}

// ---------------------------------------------------------------------------
// Kernel 3: out = res @ W_out + b_out + xs, stored transposed to (b,c,s).
// Both A (res f16) and B (WoT f16) tiles staged with async-to-LDS copies.
// ---------------------------------------------------------------------------
__global__ __launch_bounds__(256) void out_proj_kernel(
    const _Float16* __restrict__ resbuf, const _Float16* __restrict__ WoT,
    const float* __restrict__ bo, const float* __restrict__ x,
    float* __restrict__ out) {
  __shared__ __align__(16) _Float16 Al[64][40];
  __shared__ __align__(16) _Float16 BlT[128][40];

  const int m0 = blockIdx.x * 64;
  const int n0 = blockIdx.y * 128;
  const int wave = threadIdx.x >> 5, lane = threadIdx.x & 31;
  const int b = m0 >> 10, s_base = m0 & 1023;
  const int tm = wave & 3, tnb = (wave >> 2) * 4;

  v8f acc[4] = {{}, {}, {}, {}};

  for (int k0 = 0; k0 < INNER; k0 += 32) {
    __syncthreads();
    {   // A: res f16 row-major -> async b128 copy.
      int mI = threadIdx.x & 63;
      int kI = (threadIdx.x >> 6) * 8;
      async_cp16(resbuf + (size_t)(m0 + mI) * INNER + k0 + kI, &Al[mI][kI]);
    }
    {   // B: WoT f16 [n][k] -> async copy (2 ops/thread).
      int nI = threadIdx.x & 127;
      int kr = (threadIdx.x >> 7) * 16;
      async_cp32(WoT + (size_t)(n0 + nI) * DIM + k0 + kr, &BlT[nI][kr]);
    }
    wait_async0();
    __syncthreads();

    v16h a = frag_a(&Al[tm * 16][0], 40, lane);
    v16h bf[4];
#pragma unroll
    for (int j = 0; j < 4; ++j)
      bf[j] = frag_bT(&BlT[(tnb + j) * 16][0], 40, lane);
#pragma unroll
    for (int j = 0; j < 4; ++j) acc[j] = wmma16(a, bf[j], acc[j]);
  }

  const int nlane = lane & 15, hi = lane >> 4;
#pragma unroll
  for (int j = 0; j < 4; ++j) {
    int n = n0 + (tnb + j) * 16 + nlane;       // output channel c
    float bias = bo[n];
    size_t row = ((size_t)(b * DIM + n)) * SEQ + s_base + tm * 16 + 8 * hi;
    const float* xr = x + row;
    float* orow = out + row;
#pragma unroll
    for (int r = 0; r < 8; ++r) orow[r] = acc[j][r] + bias + xr[r];
  }
}

// ---------------------------------------------------------------------------
// Host launcher
// ---------------------------------------------------------------------------
extern "C" void kernel_launch(void* const* d_in, const int* in_sizes, int n_in,
                              void* d_out, int out_size, void* d_ws, size_t ws_size,
                              hipStream_t stream) {
  const float* x  = (const float*)d_in[0];   // (16, 512, 32, 32)
  const float* Wp = (const float*)d_in[1];   // (512, 1536)
  const float* bp = (const float*)d_in[2];   // (1536,)
  const float* Wo = (const float*)d_in[3];   // (512, 512)
  const float* bo = (const float*)d_in[4];   // (512,)
  float* out = (float*)d_out;                // (16, 512, 32, 32)

  const size_t elems = (size_t)BATCH * HEADS * SEQ * DIM_HEAD;  // 8,388,608
  _Float16* qbuf = (_Float16*)d_ws;
  _Float16* kbuf = qbuf + elems;
  _Float16* vbuf = kbuf + elems;
  _Float16* resb = vbuf + elems;                         // [b][s][512]
  _Float16* wpT  = resb + elems;                         // [1536][512]
  _Float16* woT  = wpT + (size_t)N_QKV * DIM;            // [512][512]

  wtrans_kernel<<<(DIM * N_QKV + 255) / 256, 256, 0, stream>>>(Wp, wpT, DIM, N_QKV);
  wtrans_kernel<<<(DIM * DIM + 255) / 256, 256, 0, stream>>>(Wo, woT, DIM, DIM);
  qkv_proj_kernel<<<dim3((BATCH * SEQ) / 64, N_QKV / 128), 256, 0, stream>>>(
      x, wpT, bp, qbuf, kbuf, vbuf);
  attn_kernel<<<dim3(BATCH * HEADS, SEQ / 64), 128, 0, stream>>>(
      qbuf, kbuf, vbuf, resb);
  out_proj_kernel<<<dim3((BATCH * SEQ) / 64, DIM / 128), 256, 0, stream>>>(
      resb, woT, bo, x, out);
}
